// LoRALinear_21388937134483
// MI455X (gfx1250) — matrix operand order repaired and checked by reading, not compiled, not run on hardware
//
#include <hip/hip_runtime.h>

// ---------------------------------------------------------------------------
// LoRA linear: out = x @ W^T + s*(x @ A^T) @ B^T
// Strategy: W' = W + s*B@A merged once into bf16 (d_ws), then one bf16 WMMA
// GEMM out = bf16(x) @ W'^T with f32 accumulation. Compute-bound problem:
// 275 GFLOP vs ~335MB HBM traffic, so keep v_wmma issue saturated.
// ---------------------------------------------------------------------------

#define IN_F    4096
#define OUT_F   4096
#define M_TOTAL 8192
#define RANK    16
#define SCALING 2.0f   // alpha/rank = 32/16

typedef __attribute__((ext_vector_type(16))) __bf16  v16bf;
typedef __attribute__((ext_vector_type(8)))  float   v8f;
typedef __attribute__((ext_vector_type(4)))  unsigned uint4x;
typedef __attribute__((ext_vector_type(2)))  unsigned uint2x;
typedef __attribute__((ext_vector_type(4)))  float    float4x;
typedef __attribute__((ext_vector_type(4)))  int      int4x;

#define GAS __attribute__((address_space(1)))   // global
#define LAS __attribute__((address_space(3)))   // LDS

// ---- optional gfx1250 async global->LDS copy path (compile-safe gating) ----
#if defined(__has_builtin)
#  if __has_builtin(__builtin_amdgcn_global_load_async_to_lds_b128)
#    define HAVE_ASYNC_LDS 1
#  endif
#  if __has_builtin(__builtin_amdgcn_s_wait_asynccnt)
#    define HAVE_WAIT_ASYNC_BUILTIN 1
#  endif
#endif

__device__ __forceinline__ void wait_async_zero() {
#if defined(HAVE_ASYNC_LDS)
#  if defined(HAVE_WAIT_ASYNC_BUILTIN)
    __builtin_amdgcn_s_wait_asynccnt(0);
#  else
    asm volatile("s_wait_asynccnt 0" ::: "memory");
#  endif
#endif
}

__device__ __forceinline__ unsigned short f32_to_bf16_rne(float f) {
    union { float f; unsigned u; } v; v.f = f;
    unsigned u = v.u;
    unsigned r = u + 0x7FFFu + ((u >> 16) & 1u);   // round-to-nearest-even
    return (unsigned short)(r >> 16);
}

// ---------------------------------------------------------------------------
// Phase 1: W'[o,k] = bf16( W[o,k] + SCALING * sum_r B[o,r]*A[r,k] )
// ---------------------------------------------------------------------------
__global__ void lora_merge_w_bf16(const float* __restrict__ W,
                                  const float* __restrict__ A,
                                  const float* __restrict__ B,
                                  unsigned short* __restrict__ Wp) {
    int id = blockIdx.x * blockDim.x + threadIdx.x;   // 0 .. 16M-1
    int o  = id >> 12;
    int k  = id & (IN_F - 1);
    float acc = W[id];
    const float* Brow = B + o * RANK;
#pragma unroll
    for (int r = 0; r < RANK; ++r)
        acc += SCALING * Brow[r] * A[r * IN_F + k];
    Wp[id] = f32_to_bf16_rne(acc);
}

// ---------------------------------------------------------------------------
// Phase 2: out[8192,4096] = bf16(x) @ W'^T, double-buffered LDS pipeline.
// Block tile 128x128, BK=32; 256 threads = 8 waves (2x4); wave computes
// 4x2 tiles of 16x16 -> 8 v_wmma_f32_16x16x32_bf16 per K-step.
// ---------------------------------------------------------------------------
#define BM 128
#define BN 128
#define BK 32
#define KST 40   // padded LDS row stride in bf16 elems (80B, conflict-free)

union Frag { v16bf v; uint4x q[2]; };

__device__ __forceinline__ void stage_tiles(const float* __restrict__ X,
                                            const unsigned short* __restrict__ Wp,
                                            unsigned short* __restrict__ sA,
                                            unsigned short* __restrict__ sB,
                                            int rowBase, int colBase, int k0, int tid) {
    const int ar = tid >> 3;          // 0..31 : A row within pass
    const int ac = (tid & 7) << 2;    // float4 column (0,4,..,28)
    const int br = tid >> 2;          // 0..63 : B row within pass
    const int bc = (tid & 3) << 3;    // 16B column in bf16 elems (0,8,16,24)

    // x tile: 128x32 f32 -> bf16 (needs VGPR conversion, 4 passes of 32 rows)
#pragma unroll
    for (int p = 0; p < 4; ++p) {
        int r = ar + p * 32;
        float4x v = *(const float4x*)(X + (size_t)(rowBase + r) * IN_F + k0 + ac);
        unsigned p0 = (unsigned)f32_to_bf16_rne(v.x) |
                      ((unsigned)f32_to_bf16_rne(v.y) << 16);
        unsigned p1 = (unsigned)f32_to_bf16_rne(v.z) |
                      ((unsigned)f32_to_bf16_rne(v.w) << 16);
        uint2x pk; pk.x = p0; pk.y = p1;
        *(uint2x*)(sA + r * KST + ac) = pk;
    }

    // W' tile: 128x32 bf16, pure byte copy -> async DMA to LDS when available
#pragma unroll
    for (int p = 0; p < 2; ++p) {
        int n = br + p * 64;
        const unsigned short* g = Wp + (size_t)(colBase + n) * IN_F + k0 + bc;
        unsigned short*       l = sB + n * KST + bc;
#if defined(HAVE_ASYNC_LDS)
        __builtin_amdgcn_global_load_async_to_lds_b128(
            (GAS int4x*)g, (LAS int4x*)l, 0, 0);
#else
        *(uint4x*)l = *(const uint4x*)g;
#endif
    }
}

__global__ __launch_bounds__(256)
void lora_gemm_bf16(const float* __restrict__ X,
                    const unsigned short* __restrict__ Wp,
                    float* __restrict__ Out) {
    __shared__ unsigned short sA[2][BM * KST];
    __shared__ unsigned short sB[2][BN * KST];

    const int tid  = threadIdx.x;
    const int wave = tid >> 5;
    const int lane = tid & 31;
    const int wm   = wave >> 2;     // 0..1  -> 64 rows each
    const int wn   = wave & 3;      // 0..3  -> 32 cols each

    const int rowBase = blockIdx.y * BM;
    const int colBase = blockIdx.x * BN;

    v8f acc[4][2];
#pragma unroll
    for (int mt = 0; mt < 4; ++mt)
#pragma unroll
        for (int nt = 0; nt < 2; ++nt)
            acc[mt][nt] = (v8f)(0.0f);

    const int laneM  = lane & 15;
    const int laneHi = lane >> 4;

    // prologue: stage first tile into buffer 0
    stage_tiles(X, Wp, sA[0], sB[0], rowBase, colBase, 0, tid);

    int cur = 0;
    for (int k0 = 0; k0 < IN_F; k0 += BK, cur ^= 1) {
        wait_async_zero();        // this wave's async copies (if any) landed
        __syncthreads();          // all waves: stage done, prev compute done

        if (k0 + BK < IN_F)
            stage_tiles(X, Wp, sA[cur ^ 1], sB[cur ^ 1],
                        rowBase, colBase, k0 + BK, tid);

        // distance-2 prefetch of the x stream (global_prefetch_b8)
        if (k0 + 2 * BK < IN_F)
            __builtin_prefetch(X + (size_t)(rowBase + (tid >> 3)) * IN_F +
                               (k0 + 2 * BK) + ((tid & 7) << 2), 0, 0);

        const unsigned short* cA = sA[cur];
        const unsigned short* cB = sB[cur];

        // A frag (16x32 bf16): lanes 0-15 row M=lane, K 0-7 & 16-23;
        //                      lanes 16-31 same rows, K 8-15 & 24-31
        Frag af[4];
#pragma unroll
        for (int mt = 0; mt < 4; ++mt) {
            int r  = wm * 64 + mt * 16 + laneM;
            int kb = laneHi * 8;
            const unsigned short* base = cA + r * KST;
            af[mt].q[0] = *(const uint4x*)(base + kb);
            af[mt].q[1] = *(const uint4x*)(base + kb + 16);
        }
        // B frag (32x16 bf16): lane holds column N, K 8h-8h+7 & 16+8h..16+8h+7 (matches A)
        Frag bfr[2];
#pragma unroll
        for (int nt = 0; nt < 2; ++nt) {
            int n  = wn * 32 + nt * 16 + laneM;
            int kh = laneHi * 8;
            const unsigned short* base = cB + n * KST + kh;
            bfr[nt].q[0] = *(const uint4x*)(base);
            bfr[nt].q[1] = *(const uint4x*)(base + 16);
        }

#pragma unroll
        for (int mt = 0; mt < 4; ++mt)
#pragma unroll
            for (int nt = 0; nt < 2; ++nt)
                acc[mt][nt] = __builtin_amdgcn_wmma_f32_16x16x32_bf16(
                    false, af[mt].v, false, bfr[nt].v,
                    (short)0, acc[mt][nt], false, false);
    }

    // epilogue: C layout lanes 0-15: M=j, lanes 16-31: M=8+j; N=lane&15.
    // Nontemporal stores: out is written once; keep W'/x resident in L2.
    const int mOff = laneHi * 8;
#pragma unroll
    for (int mt = 0; mt < 4; ++mt)
#pragma unroll
        for (int nt = 0; nt < 2; ++nt) {
            int col  = colBase + wn * 32 + nt * 16 + laneM;
            int rowb = rowBase + wm * 64 + mt * 16 + mOff;
#pragma unroll
            for (int j = 0; j < 8; ++j)
                __builtin_nontemporal_store(acc[mt][nt][j],
                                            Out + (size_t)(rowb + j) * OUT_F + col);
        }
}

// ---------------------------------------------------------------------------
extern "C" void kernel_launch(void* const* d_in, const int* in_sizes, int n_in,
                              void* d_out, int out_size, void* d_ws, size_t ws_size,
                              hipStream_t stream) {
    const float* x      = (const float*)d_in[0];
    const float* W      = (const float*)d_in[1];
    const float* lora_A = (const float*)d_in[2];
    const float* lora_B = (const float*)d_in[3];
    float* out          = (float*)d_out;

    // workspace: merged bf16 weight, 4096*4096*2 = 33.5 MB (L2-resident)
    unsigned short* Wp = (unsigned short*)d_ws;

    lora_merge_w_bf16<<<(OUT_F * IN_F) / 256, 256, 0, stream>>>(W, lora_A, lora_B, Wp);

    dim3 grid(OUT_F / BN, M_TOTAL / BM);   // col tiles fastest -> x row-block L2 reuse
    lora_gemm_bf16<<<grid, 256, 0, stream>>>(x, Wp, out);
}
